// MultiHeadLocalAttention_1683627180144
// MI455X (gfx1250) — matrix operand order repaired and checked by reading, not compiled, not run on hardware
//
#include <hip/hip_runtime.h>
#include <hip/hip_bf16.h>
#include <cstdint>
#include <cstddef>

// Problem constants
#define B_      4
#define N_      4096
#define DIM_    1024
#define HEADS_  16
#define DH_     64
#define WS_     128
#define TOK_    (B_ * N_)          // 16384 tokens

typedef __attribute__((ext_vector_type(16))) _Float16 v16h;
typedef __attribute__((ext_vector_type(8)))  float    v8f;

union HF16 { v16h h; uint32_t u[8]; };

__device__ __forceinline__ v8f v8f_zero() {
    v8f z;
#pragma unroll
    for (int i = 0; i < 8; ++i) z[i] = 0.f;
    return z;
}

__device__ __forceinline__ v8f wmma_f16(v16h a, v16h b, v8f c) {
    return __builtin_amdgcn_wmma_f32_16x16x32_f16(
        false, a, false, b, (short)0, c, false, false);
}

// ---------------- CDNA5 async / TDM helpers ----------------
// 16B-per-lane async copy global -> LDS (GV mode).  LDS byte address is the
// low 32 bits of the flat LDS pointer (aperture maps addr[31:0] to LDS).
__device__ __forceinline__ void async_ld128(uint32_t lds_byte, uint64_t gaddr) {
    asm volatile("global_load_async_to_lds_b128 %0, %1, off"
                 :: "v"(lds_byte), "v"(gaddr) : "memory");
}
__device__ __forceinline__ void wait_async0() {
    asm volatile("s_wait_asynccnt 0x0" ::: "memory");
}
__device__ __forceinline__ void wait_async3() {
    asm volatile("s_wait_asynccnt 0x3" ::: "memory");
}
__device__ __forceinline__ void wait_tensor0() {
#if __has_builtin(__builtin_amdgcn_s_wait_tensorcnt)
    __builtin_amdgcn_s_wait_tensorcnt((short)0);
#else
    asm volatile("s_wait_tensorcnt 0x0" ::: "memory");
#endif
}

#if defined(__has_builtin)
#if __has_builtin(__builtin_amdgcn_tensor_load_to_lds)
#define HAVE_TDM 1
#endif
#endif
#ifndef HAVE_TDM
#define HAVE_TDM 0
#endif

#if HAVE_TDM
typedef uint32_t u32x4 __attribute__((ext_vector_type(4)));
typedef int      i32x4 __attribute__((ext_vector_type(4)));
typedef int      i32x8 __attribute__((ext_vector_type(8)));

// 2D TDM tile load: rows x row_elems of 2-byte elements, LDS row stride =
// row bytes + 16B pad (pad_interval: 32 DWORDs, pad_amount: 4 DWORDs -> for a
// 128B row this yields a 144B (=72 half) LDS stride).
__device__ __forceinline__ void tdm_load_2d(uint32_t lds_byte, const void* gptr,
                                            uint32_t rows, uint32_t row_elems,
                                            uint32_t row_stride_elems) {
    const uint64_t ga = (uint64_t)(uintptr_t)gptr;
    u32x4 g0;
    g0[0] = 1u;                                   // count=1, user descriptor
    g0[1] = lds_byte;                             // lds_addr  (bits 63:32)
    g0[2] = (uint32_t)ga;                         // global_addr low (bits 95:64)
    g0[3] = (uint32_t)(ga >> 32) | (2u << 30);    // addr hi + type=2 (127:126)
    i32x8 g1;
    g1[0] = (1 << 16)                             // data_size = 1 (2 bytes)
          | (1 << 20)                             // pad_enable
          | (4 << 22)                             // pad_interval: 32 DWORDs
          | (3 << 25);                            // pad_amount: 4 DWORDs
    g1[1] = (int)((row_elems & 0xFFFFu) << 16);   // tensor_dim0[15:0] @63:48
    g1[2] = (int)((row_elems >> 16) | ((rows & 0xFFFFu) << 16)); // dim0 hi | tensor_dim1 lo
    g1[3] = (int)((rows >> 16) | ((row_elems & 0xFFFFu) << 16)); // dim1 hi | tile_dim0
    g1[4] = (int)(rows & 0xFFFFu);                // tile_dim1 (tile_dim2 = 0)
    g1[5] = (int)row_stride_elems;                // tensor_dim0_stride low 32
    g1[6] = 0;
    g1[7] = 0;
    i32x4 z4; z4[0] = z4[1] = z4[2] = z4[3] = 0;
    i32x8 z8;
#pragma unroll
    for (int i = 0; i < 8; ++i) z8[i] = 0;
    // 6-arg form (this toolchain): (g0, g1, g2, g3, g4, cpol)
    __builtin_amdgcn_tensor_load_to_lds(g0, g1, z4, z4, z8, 0);
}
#endif

// ---- WMMA fragment gathers from LDS (layouts per CDNA5 ISA 7.12.2) ----
__device__ __forceinline__ v16h frag_a_lds(const _Float16* tile, int stride_h,
                                           int mbase, int kbase, int lane) {
    HF16 f;
    const uint32_t* row = (const uint32_t*)(tile + (size_t)(mbase + (lane & 15)) * stride_h);
    const int klo = (kbase >> 1) + ((lane & 16) ? 4 : 0);
#pragma unroll
    for (int v = 0; v < 8; ++v)
        f.u[v] = row[klo + ((v & 4) ? 8 : 0) + (v & 3)];
    return f.h;
}
__device__ __forceinline__ v16h frag_b_lds(const _Float16* tileT, int stride_h,
                                           int nbase, int kbase, int lane) {
    HF16 f;
    const uint32_t* row = (const uint32_t*)(tileT + (size_t)(nbase + (lane & 15)) * stride_h);
    const int klo = (kbase >> 1) + ((lane & 16) ? 4 : 0);
#pragma unroll
    for (int v = 0; v < 8; ++v)
        f.u[v] = row[klo + ((v & 4) ? 8 : 0) + (v & 3)];
    return f.h;
}

// ======================= K0: weight convert f32 -> f16, TRANSPOSED ==========
// Writes W^T [N][K] so GEMM B-tiles are K-contiguous 16B async copies.
__global__ __launch_bounds__(256) void cvt_weights_kernel(
    const float* __restrict__ wqkv, const float* __restrict__ wout,
    _Float16* __restrict__ wqkvT, _Float16* __restrict__ woutT) {
    int i = blockIdx.x * 256 + threadIdx.x;
    if (i < DIM_ * 3 * HEADS_ * DH_) {
        int n = i >> 10, k = i & (DIM_ - 1);
        wqkvT[i] = (_Float16)wqkv[(size_t)k * (3 * HEADS_ * DH_) + n];
    }
    if (i < DIM_ * DIM_) {
        int n = i >> 10, k = i & (DIM_ - 1);
        woutT[i] = (_Float16)wout[(size_t)k * DIM_ + n];
    }
}

// ======================= K1: LayerNorm -> xn (f16) ===========================
__global__ __launch_bounds__(256) void layernorm_kernel(
    const float* __restrict__ x, const float* __restrict__ w,
    const float* __restrict__ b, _Float16* __restrict__ xn) {
    __shared__ float red[16];
    const int t   = blockIdx.x;
    const int tid = threadIdx.x;
    const int lane = tid & 31, wid = tid >> 5;
    const float* xr = x + (size_t)t * DIM_;
    float v[4], s = 0.f, s2 = 0.f;
#pragma unroll
    for (int i = 0; i < 4; ++i) {
        v[i] = xr[tid + 256 * i];
        s += v[i]; s2 += v[i] * v[i];
    }
#pragma unroll
    for (int m = 1; m < 32; m <<= 1) {
        s  += __shfl_xor(s,  m, 32);
        s2 += __shfl_xor(s2, m, 32);
    }
    if (lane == 0) { red[wid] = s; red[8 + wid] = s2; }
    __syncthreads();
    float S = 0.f, S2 = 0.f;
#pragma unroll
    for (int wv = 0; wv < 8; ++wv) { S += red[wv]; S2 += red[8 + wv]; }
    const float mu  = S * (1.f / DIM_);
    const float var = S2 * (1.f / DIM_) - mu * mu;
    const float rs  = rsqrtf(var + 1e-5f);
#pragma unroll
    for (int i = 0; i < 4; ++i) {
        int c = tid + 256 * i;
        xn[(size_t)t * DIM_ + c] = (_Float16)((v[i] - mu) * rs * w[c] + b[c]);
    }
}

// ======================= K2: gates = sigmoid(xn @ w_gate + b) ================
__global__ __launch_bounds__(256) void gate_kernel(
    const _Float16* __restrict__ xn, const float* __restrict__ wg,
    const float* __restrict__ bg, float* __restrict__ gates) {
    const int idx = blockIdx.x * 256 + threadIdx.x;
    const int t = idx >> 4, h = idx & 15;
    const _Float16* xr = xn + (size_t)t * DIM_;
    float s = 0.f;
    for (int k = 0; k < DIM_; k += 8) {
        union { uint4 q; _Float16 hh[8]; } u;
        u.q = *(const uint4*)(xr + k);
#pragma unroll
        for (int i = 0; i < 8; ++i)
            s += (float)u.hh[i] * wg[(k + i) * HEADS_ + h];
    }
    gates[idx] = 1.f / (1.f + __expf(-(s + bg[h])));
}

// ======================= async double-buffered WMMA GEMM mainloop ============
// C tile 128x64 per block (8 waves, 32x32 each).  A: [M][K] f16.  BT: [N][K] f16.
constexpr int SAB_STR     = 40;                     // halves; 80B = 5x16B rows
constexpr int ABUF_HALVES = 128 * SAB_STR;          // 10240 B
constexpr int BBUF_HALVES = 64 * SAB_STR;           //  5120 B
constexpr int BUF_HALVES  = ABUF_HALVES + BBUF_HALVES;
constexpr int BUF_BYTES   = BUF_HALVES * 2;         // 15360 B per buffer

__device__ __forceinline__ void gemm_mainloop_async(
    const _Float16* __restrict__ A, int lda,
    const _Float16* __restrict__ BT, int ldbt,
    int blockM, int blockN, int K,
    _Float16* sbase, v8f acc[2][2], int tid) {
    const int lane = tid & 31, wid = tid >> 5;
    const int m0 = (wid >> 1) * 32, n0 = (wid & 1) * 32;
    const int rl = tid >> 1, ha = (tid & 1) * 16;   // A: 2 thr/row, 32B each
    const int nb = tid >> 2, kb = (tid & 3) * 8;    // BT: 4 thr/row, 16B each
    const uint64_t gA = (uint64_t)(uintptr_t)(A  + (size_t)(blockM + rl) * lda  + ha);
    const uint64_t gB = (uint64_t)(uintptr_t)(BT + (size_t)(blockN + nb) * ldbt + kb);
    const uint32_t lA = (uint32_t)(uintptr_t)(sbase + (size_t)rl * SAB_STR + ha);
    const uint32_t lB = (uint32_t)(uintptr_t)(sbase + ABUF_HALVES + (size_t)nb * SAB_STR + kb);
    const int nk = K / 32;
    // prologue: stage K-chunk 0 into buffer 0
    async_ld128(lA,      gA);
    async_ld128(lA + 16, gA + 16);
    async_ld128(lB,      gB);
    for (int kk = 0; kk < nk; ++kk) {
        if (kk + 1 < nk) {   // stage next chunk into the other buffer
            const uint32_t no = (uint32_t)(((kk + 1) & 1) * BUF_BYTES);
            const uint64_t ga = gA + (uint64_t)(kk + 1) * 64;
            const uint64_t gb = gB + (uint64_t)(kk + 1) * 64;
            async_ld128(lA + no,      ga);
            async_ld128(lA + no + 16, ga + 16);
            async_ld128(lB + no,      gb);
            if (kk + 4 < nk)
                __builtin_prefetch(A + (size_t)(blockM + rl) * lda + (kk + 4) * 32 + ha, 0, 0);
            wait_async3();   // current buffer's 3 DMAs done; next 3 in flight
        } else {
            wait_async0();
        }
        __syncthreads();
        const _Float16* cA = sbase + (size_t)(kk & 1) * BUF_HALVES;
        const _Float16* cB = cA + ABUF_HALVES;
        v16h a0 = frag_a_lds(cA, SAB_STR, m0,      0, lane);
        v16h a1 = frag_a_lds(cA, SAB_STR, m0 + 16, 0, lane);
        v16h b0 = frag_b_lds(cB, SAB_STR, n0,      0, lane);
        v16h b1 = frag_b_lds(cB, SAB_STR, n0 + 16, 0, lane);
        acc[0][0] = wmma_f16(a0, b0, acc[0][0]);
        acc[0][1] = wmma_f16(a0, b1, acc[0][1]);
        acc[1][0] = wmma_f16(a1, b0, acc[1][0]);
        acc[1][1] = wmma_f16(a1, b1, acc[1][1]);
        __syncthreads();     // all reads done before this buffer is re-filled
    }
}

// ======================= K3: QKV GEMM + RoPE epilogue ========================
__global__ __launch_bounds__(256) void qkv_rope_kernel(
    const _Float16* __restrict__ xn, const _Float16* __restrict__ wqkvT,
    _Float16* __restrict__ qh, _Float16* __restrict__ kh, _Float16* __restrict__ vh) {
    extern __shared__ char smem[];
    _Float16* sbase = (_Float16*)smem;
    float*    sC    = (float*)smem;                 // reused after mainloop
    const int tid = threadIdx.x, lane = tid & 31, wid = tid >> 5;
    const int blockM = blockIdx.x * 128, blockN = blockIdx.y * 64;

    v8f acc[2][2];
    acc[0][0] = v8f_zero(); acc[0][1] = v8f_zero();
    acc[1][0] = v8f_zero(); acc[1][1] = v8f_zero();
    gemm_mainloop_async(xn, DIM_, wqkvT, DIM_, blockM, blockN, DIM_, sbase, acc, tid);

    const int m0 = (wid >> 1) * 32, n0 = (wid & 1) * 32;
    const int mhi = (lane & 16) ? 8 : 0;
#pragma unroll
    for (int i = 0; i < 2; ++i)
#pragma unroll
        for (int j = 0; j < 2; ++j)
#pragma unroll
            for (int r = 0; r < 8; ++r)
                sC[(size_t)(m0 + i * 16 + r + mhi) * 68 + n0 + j * 16 + (lane & 15)] =
                    acc[i][j][r];
    __syncthreads();

    const int mat  = blockN >> 10;          // 0=q 1=k 2=v
    const int head = (blockN >> 6) & 15;
    for (int e = tid; e < 128 * 64; e += 256) {
        const int row = e >> 6, col = e & 63;
        const int t = blockM + row;
        const int pos = t & (N_ - 1), bb = t >> 12;
        const int d = col;
        const size_t dst = ((size_t)(bb * HEADS_ + head) * N_ + pos) * DH_ + d;
        const float val = sC[(size_t)row * 68 + col];
        if (mat == 2) {
            vh[dst] = (_Float16)val;
        } else {
            const int jf = d & 31;
            const float invf = __powf(10000.f, -(float)jf * (1.f / 32.f));
            float sn, cs;
            __sincosf((float)pos * invf, &sn, &cs);
            const float partner = sC[(size_t)row * 68 + ((d < 32) ? col + 32 : col - 32)];
            const float rot = (d < 32) ? -partner : partner;
            const float rv  = val * cs + rot * sn;
            if (mat == 0) qh[dst] = (_Float16)(rv * 0.125f);
            else          kh[dst] = (_Float16)rv;
        }
    }
}

// ======================= K4: local windowed attention ========================
constexpr int QS_STR = 72;     // q tile  128 x 64, 144B rows (TDM-padded)
constexpr int KS_STR = 72;     // k tile  256 x 64, 144B rows (TDM-padded)
constexpr int VT_STR = 264;    // v^T     64 x 256
constexpr int AT_STR = 264;    // attn    16 x 256 per wave

__global__ __launch_bounds__(256) void local_attn_kernel(
    const _Float16* __restrict__ qh, const _Float16* __restrict__ kh,
    const _Float16* __restrict__ vh, const float* __restrict__ gates,
    _Float16* __restrict__ ao) {
    extern __shared__ char smem[];
    _Float16* qs  = (_Float16*)smem;
    _Float16* ks  = qs  + 128 * QS_STR;
    _Float16* vts = ks  + 256 * KS_STR;
    _Float16* at  = vts + 64  * VT_STR;
    const int tid = threadIdx.x, lane = tid & 31, wid = tid >> 5;
    const int bh = blockIdx.x, win = blockIdx.y;

    // ---- zero-pad first half of k context when win==0 (t_k < 0) ----
    if (win == 0 && tid < 128) {
        uint4 z; z.x = z.y = z.z = z.w = 0u;
        uint4* d = (uint4*)(ks + (size_t)tid * KS_STR);
#pragma unroll
        for (int i = 0; i < 8; ++i) d[i] = z;
    }

    // ---- q and k context tiles via Tensor Data Mover (fallback: async) ----
#if HAVE_TDM
    if (wid == 0) {
        tdm_load_2d((uint32_t)(uintptr_t)qs,
                    qh + ((size_t)bh * N_ + win * WS_) * DH_, WS_, DH_, DH_);
        if (win > 0)
            tdm_load_2d((uint32_t)(uintptr_t)ks,
                        kh + ((size_t)bh * N_ + (win - 1) * WS_) * DH_,
                        2 * WS_, DH_, DH_);
        else
            tdm_load_2d((uint32_t)(uintptr_t)(ks + (size_t)WS_ * KS_STR),
                        kh + ((size_t)bh * N_) * DH_, WS_, DH_, DH_);
        wait_tensor0();
    }
#else
    {   // q: 2 threads per row, 64B each
        const int row = tid >> 1, hoff = (tid & 1) * 32;
        const uint64_t g = (uint64_t)(uintptr_t)
            (qh + ((size_t)bh * N_ + win * WS_ + row) * DH_ + hoff);
        const uint32_t l = (uint32_t)(uintptr_t)(qs + (size_t)row * QS_STR + hoff);
#pragma unroll
        for (int i = 0; i < 4; ++i) async_ld128(l + 16 * i, g + 16 * i);
    }
    {   // k: one row per thread (valid rows only; invalid ones zeroed above)
        const int p = win * WS_ - WS_ + tid;
        if (p >= 0) {
            const uint64_t g = (uint64_t)(uintptr_t)(kh + ((size_t)bh * N_ + p) * DH_);
            const uint32_t l = (uint32_t)(uintptr_t)(ks + (size_t)tid * KS_STR);
#pragma unroll
            for (int i = 0; i < 8; ++i) async_ld128(l + 16 * i, g + 16 * i);
        }
    }
    wait_async0();
#endif

    // ---- v context, transposed into [d][key] ----
    {
        const int j = tid;
        const int p = win * WS_ - WS_ + j;
        union { uint4 q[8]; _Float16 h[64]; } vr;
        if (p >= 0) {
            const uint4* s = (const uint4*)(vh + ((size_t)bh * N_ + p) * DH_);
#pragma unroll
            for (int i = 0; i < 8; ++i) vr.q[i] = s[i];
        } else {
#pragma unroll
            for (int i = 0; i < 8; ++i) { vr.q[i].x = vr.q[i].y = vr.q[i].z = vr.q[i].w = 0u; }
        }
#pragma unroll
        for (int d = 0; d < DH_; ++d) vts[(size_t)d * VT_STR + j] = vr.h[d];
    }
    __syncthreads();

    // ---- sim = q @ k^T  (16 rows x 256 keys per wave) ----
    v8f sim[16];
#pragma unroll
    for (int nt = 0; nt < 16; ++nt) sim[nt] = v8f_zero();
    const v16h aq0 = frag_a_lds(qs, QS_STR, wid * 16, 0,  lane);
    const v16h aq1 = frag_a_lds(qs, QS_STR, wid * 16, 32, lane);
#pragma unroll
    for (int nt = 0; nt < 16; ++nt) {
        v16h bk0 = frag_b_lds(ks, KS_STR, nt * 16, 0,  lane);
        sim[nt] = wmma_f16(aq0, bk0, sim[nt]);
        v16h bk1 = frag_b_lds(ks, KS_STR, nt * 16, 32, lane);
        sim[nt] = wmma_f16(aq1, bk1, sim[nt]);
    }

    // ---- mask + softmax ----
    const int mhi = (lane & 16) ? 8 : 0;
    const int jn  = lane & 15;
    float rmax[8], rsum[8];
#pragma unroll
    for (int r = 0; r < 8; ++r) {
        const int rw = wid * 16 + r + mhi;
        float mx = -3.0e38f;
#pragma unroll
        for (int nt = 0; nt < 16; ++nt) {
            const int j = nt * 16 + jn;
            const bool valid = (j >= rw) && (j <= rw + WS_) && (win > 0 || j >= WS_);
            const float sv = valid ? sim[nt][r] : -3.0e38f;
            sim[nt][r] = sv;
            mx = fmaxf(mx, sv);
        }
#pragma unroll
        for (int m = 1; m < 16; m <<= 1) mx = fmaxf(mx, __shfl_xor(mx, m, 32));
        rmax[r] = mx;
    }
#pragma unroll
    for (int r = 0; r < 8; ++r) {
        float sum = 0.f;
#pragma unroll
        for (int nt = 0; nt < 16; ++nt) {
            const float e = __expf(sim[nt][r] - rmax[r]);
            sim[nt][r] = e;
            sum += e;
        }
#pragma unroll
        for (int m = 1; m < 16; m <<= 1) sum += __shfl_xor(sum, m, 32);
        rsum[r] = 1.f / sum;
    }

    // ---- attn probs (f16) to this wave's private LDS strip ----
    _Float16* myat = at + (size_t)wid * 16 * AT_STR;
#pragma unroll
    for (int nt = 0; nt < 16; ++nt)
#pragma unroll
        for (int r = 0; r < 8; ++r)
            myat[(size_t)(r + mhi) * AT_STR + nt * 16 + jn] =
                (_Float16)(sim[nt][r] * rsum[r]);

    // ---- out = attn @ v  (16 x 64 per wave, K = 256) ----
    v8f o[4];
#pragma unroll
    for (int dt = 0; dt < 4; ++dt) o[dt] = v8f_zero();
#pragma unroll
    for (int kkc = 0; kkc < 8; ++kkc) {
        v16h ap = frag_a_lds(myat, AT_STR, 0, kkc * 32, lane);
#pragma unroll
        for (int dt = 0; dt < 4; ++dt) {
            v16h bv = frag_b_lds(vts, VT_STR, dt * 16, kkc * 32, lane);
            o[dt] = wmma_f16(ap, bv, o[dt]);
        }
    }

    // ---- gate + store f16 token-major [t][head*64 + d] ----
    const int bb = bh >> 4, head = bh & 15;
#pragma unroll
    for (int r = 0; r < 8; ++r) {
        const int rw = wid * 16 + r + mhi;
        const size_t t = (size_t)bb * N_ + win * WS_ + rw;
        const float g = gates[t * HEADS_ + head];
#pragma unroll
        for (int dt = 0; dt < 4; ++dt) {
            const int d = dt * 16 + jn;
            ao[t * (HEADS_ * DH_) + head * DH_ + d] = (_Float16)(o[dt][r] * g);
        }
    }
}

// ======================= K5: output projection GEMM ==========================
__global__ __launch_bounds__(256) void out_gemm_kernel(
    const _Float16* __restrict__ ao, const _Float16* __restrict__ woutT,
    float* __restrict__ out) {
    extern __shared__ char smem[];
    _Float16* sbase = (_Float16*)smem;
    const int tid = threadIdx.x, lane = tid & 31, wid = tid >> 5;
    const int blockM = blockIdx.x * 128, blockN = blockIdx.y * 64;

    v8f acc[2][2];
    acc[0][0] = v8f_zero(); acc[0][1] = v8f_zero();
    acc[1][0] = v8f_zero(); acc[1][1] = v8f_zero();
    gemm_mainloop_async(ao, DIM_, woutT, DIM_, blockM, blockN, DIM_, sbase, acc, tid);

    const int m0 = (wid >> 1) * 32, n0 = (wid & 1) * 32;
    const int mhi = (lane & 16) ? 8 : 0;
#pragma unroll
    for (int i = 0; i < 2; ++i)
#pragma unroll
        for (int j = 0; j < 2; ++j)
#pragma unroll
            for (int r = 0; r < 8; ++r)
                out[(size_t)(blockM + m0 + i * 16 + r + mhi) * DIM_ +
                    blockN + n0 + j * 16 + (lane & 15)] = acc[i][j][r];
}

// ======================= host launcher =======================================
extern "C" void kernel_launch(void* const* d_in, const int* in_sizes, int n_in,
                              void* d_out, int out_size, void* d_ws, size_t ws_size,
                              hipStream_t stream) {
    (void)in_sizes; (void)n_in; (void)out_size; (void)ws_size;
    const float* x      = (const float*)d_in[0];
    const float* ln_w   = (const float*)d_in[1];
    const float* ln_b   = (const float*)d_in[2];
    const float* w_qkv  = (const float*)d_in[3];
    const float* w_gate = (const float*)d_in[4];
    const float* b_gate = (const float*)d_in[5];
    const float* w_out  = (const float*)d_in[6];
    float* out = (float*)d_out;

    char* ws = (char*)d_ws;
    size_t off = 0;
    auto alloc = [&](size_t bytes) -> char* {
        char* p = ws + off;
        off += (bytes + 255) & ~(size_t)255;
        return p;
    };
    _Float16* xn_h    = (_Float16*)alloc((size_t)TOK_ * DIM_ * 2);
    _Float16* wqkv_hT = (_Float16*)alloc((size_t)DIM_ * 3 * HEADS_ * DH_ * 2);
    _Float16* wout_hT = (_Float16*)alloc((size_t)DIM_ * DIM_ * 2);
    _Float16* q_h     = (_Float16*)alloc((size_t)TOK_ * DIM_ * 2);
    _Float16* k_h     = (_Float16*)alloc((size_t)TOK_ * DIM_ * 2);
    _Float16* v_h     = (_Float16*)alloc((size_t)TOK_ * DIM_ * 2);
    float*    gates   = (float*)   alloc((size_t)TOK_ * HEADS_ * 4);
    _Float16* ao_h    = (_Float16*)alloc((size_t)TOK_ * DIM_ * 2);

    const int GEMM_SMEM = 128 * 68 * 4;   // epilogue f32 tile (34816 B) > 2*15360
    const int ATTN_SMEM = (128 * QS_STR + 256 * KS_STR + 64 * VT_STR + 128 * AT_STR) * 2;

    (void)hipFuncSetAttribute(reinterpret_cast<const void*>(qkv_rope_kernel),
                              hipFuncAttributeMaxDynamicSharedMemorySize, GEMM_SMEM);
    (void)hipFuncSetAttribute(reinterpret_cast<const void*>(out_gemm_kernel),
                              hipFuncAttributeMaxDynamicSharedMemorySize, GEMM_SMEM);
    (void)hipFuncSetAttribute(reinterpret_cast<const void*>(local_attn_kernel),
                              hipFuncAttributeMaxDynamicSharedMemorySize, ATTN_SMEM);

    cvt_weights_kernel<<<(DIM_ * 3 * HEADS_ * DH_ + 255) / 256, 256, 0, stream>>>(
        w_qkv, w_out, wqkv_hT, wout_hT);
    layernorm_kernel<<<TOK_, 256, 0, stream>>>(x, ln_w, ln_b, xn_h);
    gate_kernel<<<(TOK_ * HEADS_) / 256, 256, 0, stream>>>(xn_h, w_gate, b_gate, gates);
    qkv_rope_kernel<<<dim3(TOK_ / 128, (3 * HEADS_ * DH_) / 64), 256, GEMM_SMEM, stream>>>(
        xn_h, wqkv_hT, q_h, k_h, v_h);
    local_attn_kernel<<<dim3(B_ * HEADS_, N_ / WS_), 256, ATTN_SMEM, stream>>>(
        q_h, k_h, v_h, gates, ao_h);
    out_gemm_kernel<<<dim3(TOK_ / 128, DIM_ / 64), 256, GEMM_SMEM, stream>>>(
        ao_h, wout_hT, out);
}
